// GCN_18176301597601
// MI455X (gfx1250) — matrix operand order repaired and checked by reading, not compiled, not run on hardware
//
#include <hip/hip_runtime.h>

// ---------------- problem constants ----------------
#define NN     50000     // nodes
#define NE     800000    // edges
#define FIN    500       // input features
#define FH     128       // hidden features
#define FO     47        // output features
#define FOP    48        // padded output features (3x16 WMMA tiles)
#define KPAD1  512       // FIN padded to multiple of 32

typedef __attribute__((ext_vector_type(16))) __bf16 v16bf;
typedef __attribute__((ext_vector_type(8)))  float  v8f;

// ---------------- bf16 helpers (RNE) ----------------
__device__ __forceinline__ unsigned short f2bfbits(float f) {
  unsigned u = __builtin_bit_cast(unsigned, f);
  return (unsigned short)((u + 0x7FFFu + ((u >> 16) & 1u)) >> 16);
}
__device__ __forceinline__ float bfbits2f(unsigned short b) {
  unsigned u = ((unsigned)b) << 16;
  return __builtin_bit_cast(float, u);
}
__device__ __forceinline__ __bf16 bits2bf(unsigned short b) {
  return __builtin_bit_cast(__bf16, b);
}

// split 16 floats into bf16 hi/lo fragment pair (A-matrix fragment)
__device__ __forceinline__ void cvt16(const float* __restrict__ f, v16bf& ah, v16bf& al) {
#pragma unroll
  for (int e = 0; e < 16; ++e) {
    unsigned short hb = f2bfbits(f[e]);
    ah[e] = bits2bf(hb);
    al[e] = bits2bf(f2bfbits(f[e] - bfbits2f(hb)));
  }
}

// A fragment (16x32 bf16 layout): lane L<16 holds row M=L, K = kbase+{0..7,16..23};
// lane L>=16 holds row M=L-16, K = kbase+{8..15,24..31}.
__device__ __forceinline__ void load_afrag_full(const float* __restrict__ rowp,
                                                int kbase, int lane,
                                                v16bf& ah, v16bf& al) {
  const int h = (lane >> 4) * 8;
  const float4* p0 = reinterpret_cast<const float4*>(rowp + kbase + h);
  const float4* p1 = reinterpret_cast<const float4*>(rowp + kbase + 16 + h);
  float4 x0 = p0[0], x1 = p0[1], y0 = p1[0], y1 = p1[1];
  float f[16] = {x0.x, x0.y, x0.z, x0.w, x1.x, x1.y, x1.z, x1.w,
                 y0.x, y0.y, y0.z, y0.w, y1.x, y1.y, y1.z, y1.w};
  cvt16(f, ah, al);
}

// K-tail variant: clamp address + zero via multiply (cndmask, no branches)
__device__ __forceinline__ void load_afrag_tail(const float* __restrict__ rowp,
                                                int kbase, int lane, int kmax,
                                                v16bf& ah, v16bf& al) {
  const int h = (lane >> 4) * 8;
  float f[16];
#pragma unroll
  for (int e = 0; e < 8; ++e) {
    int k0 = kbase + h + e;
    int k1 = kbase + 16 + h + e;
    float m0 = (k0 < kmax) ? 1.f : 0.f;
    float m1 = (k1 < kmax) ? 1.f : 0.f;
    int c0 = (k0 < kmax) ? k0 : (kmax - 1);
    int c1 = (k1 < kmax) ? k1 : (kmax - 1);
    f[e]     = rowp[c0] * m0;
    f[e + 8] = rowp[c1] * m1;
  }
  cvt16(f, ah, al);
}

// B fragment (32x16 bf16): lane L<16 holds col N=L, K = kbase+{0..7,16..23};
// lane L>=16 holds col N=L-16, K = kbase+{8..15,24..31}. Wt is column-major bf16 bits
// (Wt[col*ldk + k]) so each lane reads 2 x 8 contiguous bf16 = 2 x b128.
__device__ __forceinline__ v16bf load_bfrag(const unsigned short* __restrict__ Wt,
                                            int col, int ldk, int kbase, int lane) {
  const unsigned short* p = Wt + (size_t)col * ldk + kbase + (lane >> 4) * 8;
  union { uint4 q[2]; v16bf v; } u;
  u.q[0] = *reinterpret_cast<const uint4*>(p);
  u.q[1] = *reinterpret_cast<const uint4*>(p + 16);
  return u.v;
}

// bf16x3 accumulate: acc += Ah*Bh + Ah*Bl + Al*Bh  (fp32 accumulation)
__device__ __forceinline__ v8f wmma3(v16bf ah, v16bf al, v16bf bh, v16bf bl, v8f acc) {
  acc = __builtin_amdgcn_wmma_f32_16x16x32_bf16(false, ah, false, bh, (short)0, acc, false, false);
  acc = __builtin_amdgcn_wmma_f32_16x16x32_bf16(false, ah, false, bl, (short)0, acc, false, false);
  acc = __builtin_amdgcn_wmma_f32_16x16x32_bf16(false, al, false, bh, (short)0, acc, false, false);
  return acc;
}

// ---------------- weight prep: transpose + pad + bf16 hi/lo split ----------------
__global__ void gcn_prep_weights(const float* __restrict__ W1, const float* __restrict__ W2,
                                 unsigned short* __restrict__ W1tH, unsigned short* __restrict__ W1tL,
                                 unsigned short* __restrict__ W2tH, unsigned short* __restrict__ W2tL) {
  int tid = blockIdx.x * blockDim.x + threadIdx.x;
  const int n1 = FH * KPAD1;    // 65536
  const int n2 = FOP * FH;      // 6144
  if (tid < n1) {
    int c = tid / KPAD1, k = tid - c * KPAD1;
    float w = (k < FIN) ? W1[(size_t)k * FH + c] : 0.f;
    unsigned short hb = f2bfbits(w);
    W1tH[tid] = hb;
    W1tL[tid] = f2bfbits(w - bfbits2f(hb));
  } else if (tid < n1 + n2) {
    int t = tid - n1;
    int c = t / FH, k = t - c * FH;
    float w = (c < FO) ? W2[(size_t)k * FO + c] : 0.f;
    unsigned short hb = f2bfbits(w);
    W2tH[t] = hb;
    W2tL[t] = f2bfbits(w - bfbits2f(hb));
  }
}

// ---------------- degree / norm ----------------
__global__ void gcn_deg_init(float* __restrict__ deg) {
  int i = blockIdx.x * blockDim.x + threadIdx.x;
  if (i < NN) deg[i] = 1.f;   // self-loop
}
__global__ void gcn_deg_accum(const int* __restrict__ ei, float* __restrict__ deg) {
  int e = blockIdx.x * blockDim.x + threadIdx.x;
  if (e < NE) atomicAdd(&deg[ei[NE + e]], 1.f);   // edge_index[1] = dst
}
__global__ void gcn_deg_finalize(float* __restrict__ deg) {
  int i = blockIdx.x * blockDim.x + threadIdx.x;
  if (i < NN) deg[i] = rsqrtf(deg[i]);            // deg >= 1 always
}

// ---------------- GEMM1: Hpre[N,128] = X[N,500] @ W1 ----------------
__global__ void __launch_bounds__(256)
gcn_gemm1_wmma(const float* __restrict__ X,
               const unsigned short* __restrict__ W1tH,
               const unsigned short* __restrict__ W1tL,
               float* __restrict__ Hpre) {
  const int lane = threadIdx.x & 31;
  const int wid  = (blockIdx.x * 256 + threadIdx.x) >> 5;
  if (wid >= NN / 16) return;                       // wave-uniform
  const int m0   = wid * 16;
  const int bcol = lane & 15;
  const float* rowp = X + (size_t)(m0 + bcol) * FIN;

  v8f zero = {};
  v8f acc[8];
#pragma unroll
  for (int n = 0; n < 8; ++n) acc[n] = zero;

#pragma unroll 1
  for (int ks = 0; ks < 15; ++ks) {                 // K = 0..479, all < 500
    const int kbase = ks * 32;
    v16bf ah, al;
    load_afrag_full(rowp, kbase, lane, ah, al);
#pragma unroll
    for (int n = 0; n < 8; ++n) {
      v16bf bh = load_bfrag(W1tH, n * 16 + bcol, KPAD1, kbase, lane);
      v16bf bl = load_bfrag(W1tL, n * 16 + bcol, KPAD1, kbase, lane);
      acc[n] = wmma3(ah, al, bh, bl, acc[n]);
    }
  }
  { // tail: K = 480..511, mask K >= 500 (W planes already zero-padded there)
    const int kbase = 480;
    v16bf ah, al;
    load_afrag_tail(rowp, kbase, lane, FIN, ah, al);
#pragma unroll
    for (int n = 0; n < 8; ++n) {
      v16bf bh = load_bfrag(W1tH, n * 16 + bcol, KPAD1, kbase, lane);
      v16bf bl = load_bfrag(W1tL, n * 16 + bcol, KPAD1, kbase, lane);
      acc[n] = wmma3(ah, al, bh, bl, acc[n]);
    }
  }
  // C/D layout: VGPR j = row (j + 8*(lane>=16)), col = lane&15
  const int rbase = m0 + ((lane >> 4) * 8);
#pragma unroll
  for (int n = 0; n < 8; ++n)
#pragma unroll
    for (int j = 0; j < 8; ++j)
      Hpre[(size_t)(rbase + j) * FH + n * 16 + bcol] = acc[n][j];
}

// ---------------- GEMM2: H2[N,48] = H1[N,128] @ W2(pad) ----------------
__global__ void __launch_bounds__(256)
gcn_gemm2_wmma(const float* __restrict__ H1,
               const unsigned short* __restrict__ W2tH,
               const unsigned short* __restrict__ W2tL,
               float* __restrict__ H2) {
  const int lane = threadIdx.x & 31;
  const int wid  = (blockIdx.x * 256 + threadIdx.x) >> 5;
  if (wid >= NN / 16) return;
  const int m0   = wid * 16;
  const int bcol = lane & 15;
  const float* rowp = H1 + (size_t)(m0 + bcol) * FH;

  v8f zero = {};
  v8f acc[3];
#pragma unroll
  for (int n = 0; n < 3; ++n) acc[n] = zero;

#pragma unroll
  for (int ks = 0; ks < 4; ++ks) {                  // K = 128 exactly
    const int kbase = ks * 32;
    v16bf ah, al;
    load_afrag_full(rowp, kbase, lane, ah, al);
#pragma unroll
    for (int n = 0; n < 3; ++n) {
      v16bf bh = load_bfrag(W2tH, n * 16 + bcol, FH, kbase, lane);
      v16bf bl = load_bfrag(W2tL, n * 16 + bcol, FH, kbase, lane);
      acc[n] = wmma3(ah, al, bh, bl, acc[n]);
    }
  }
  const int rbase = m0 + ((lane >> 4) * 8);
#pragma unroll
  for (int n = 0; n < 3; ++n)
#pragma unroll
    for (int j = 0; j < 8; ++j)
      H2[(size_t)(rbase + j) * FOP + n * 16 + bcol] = acc[n][j];
}

// ---------------- zero init ----------------
__global__ void gcn_zero4(float* __restrict__ p, int n4) {
  int i = blockIdx.x * blockDim.x + threadIdx.x;
  if (i < n4) reinterpret_cast<float4*>(p)[i] = make_float4(0.f, 0.f, 0.f, 0.f);
}

// ---------------- scatter-add aggregation, layer 1 (128 feats) ----------------
// message m < NE: edge (src=ei[m], dst=ei[NE+m]); m >= NE: self-loop node m-NE.
__global__ void gcn_scatter1(const int* __restrict__ ei, const float* __restrict__ dinv,
                             const float* __restrict__ Hpre, float* __restrict__ Agg) {
  const int lane = threadIdx.x & 31;
  const int m = (blockIdx.x * blockDim.x + threadIdx.x) >> 5;
  if (m >= NE + NN) return;
  int s, d;
  if (m < NE) { s = ei[m]; d = ei[NE + m]; }
  else        { s = m - NE; d = s; }
  const float nrm = dinv[s] * dinv[d];
  const float4 v = *reinterpret_cast<const float4*>(Hpre + (size_t)s * FH + lane * 4);
  float* o = Agg + (size_t)d * FH + lane * 4;
  atomicAdd(o + 0, v.x * nrm);
  atomicAdd(o + 1, v.y * nrm);
  atomicAdd(o + 2, v.z * nrm);
  atomicAdd(o + 3, v.w * nrm);
}

// ---------------- bias + relu (in place on Agg1 -> H1) ----------------
__global__ void gcn_bias_relu(float* __restrict__ H, const float* __restrict__ b1) {
  int i = blockIdx.x * blockDim.x + threadIdx.x;
  if (i < NN * FH) {
    float v = H[i] + b1[i & (FH - 1)];
    H[i] = v > 0.f ? v : 0.f;
  }
}

// ---------------- scatter-add aggregation, layer 2 (48 padded feats) ----------------
__global__ void gcn_scatter2(const int* __restrict__ ei, const float* __restrict__ dinv,
                             const float* __restrict__ H2, float* __restrict__ Agg) {
  const int lane = threadIdx.x & 31;
  const int m = (blockIdx.x * blockDim.x + threadIdx.x) >> 5;
  if (m >= NE + NN) return;
  int s, d;
  if (m < NE) { s = ei[m]; d = ei[NE + m]; }
  else        { s = m - NE; d = s; }
  const float nrm = dinv[s] * dinv[d];
  const float* r = H2 + (size_t)s * FOP;
  float* o = Agg + (size_t)d * FOP;
  atomicAdd(o + lane, r[lane] * nrm);
  if (lane < 16) atomicAdd(o + 32 + lane, r[32 + lane] * nrm);
}

// ---------------- bias + log_softmax (wave per node) ----------------
__global__ void gcn_logsoftmax(const float* __restrict__ Agg2, const float* __restrict__ b2,
                               float* __restrict__ out) {
  const int lane = threadIdx.x & 31;
  const int node = (blockIdx.x * blockDim.x + threadIdx.x) >> 5;
  if (node >= NN) return;
  const float* r = Agg2 + (size_t)node * FOP;
  float v0 = r[lane] + b2[lane];                       // lane 0..31 < 47
  bool has1 = lane < (FO - 32);                        // lanes 0..14
  float v1 = has1 ? (r[32 + lane] + b2[32 + lane]) : -3.402823e38f;
  float mx = fmaxf(v0, v1);
#pragma unroll
  for (int off = 16; off > 0; off >>= 1) mx = fmaxf(mx, __shfl_xor(mx, off, 32));
  float s = __expf(v0 - mx) + (has1 ? __expf(v1 - mx) : 0.f);
#pragma unroll
  for (int off = 16; off > 0; off >>= 1) s += __shfl_xor(s, off, 32);
  float lse = mx + __logf(s);
  out[(size_t)node * FO + lane] = v0 - lse;
  if (has1) out[(size_t)node * FO + 32 + lane] = v1 - lse;
}

// ---------------- host launcher ----------------
extern "C" void kernel_launch(void* const* d_in, const int* in_sizes, int n_in,
                              void* d_out, int out_size, void* d_ws, size_t ws_size,
                              hipStream_t stream) {
  const float* X  = (const float*)d_in[0];
  const float* W1 = (const float*)d_in[1];
  const float* b1 = (const float*)d_in[2];
  const float* W2 = (const float*)d_in[3];
  const float* b2 = (const float*)d_in[4];
  const int*   ei = (const int*)d_in[5];   // [2, NE] flattened, int32
  float* out = (float*)d_out;
  char* ws = (char*)d_ws;

  auto up = [](size_t x) { return (x + 255) & ~(size_t)255; };
  size_t o = 0;
  float*          dinv = (float*)(ws + o);          o = up(o + (size_t)NN * 4);
  unsigned short* W1tH = (unsigned short*)(ws + o); o = up(o + (size_t)FH * KPAD1 * 2);
  unsigned short* W1tL = (unsigned short*)(ws + o); o = up(o + (size_t)FH * KPAD1 * 2);
  unsigned short* W2tH = (unsigned short*)(ws + o); o = up(o + (size_t)FOP * FH * 2);
  unsigned short* W2tL = (unsigned short*)(ws + o); o = up(o + (size_t)FOP * FH * 2);
  float*          Hpre = (float*)(ws + o);          o = up(o + (size_t)NN * FH * 4);
  float*          Agg1 = (float*)(ws + o);          o = up(o + (size_t)NN * FH * 4);
  float*          H2   = (float*)(ws + o);          o = up(o + (size_t)NN * FOP * 4);
  float*          Agg2 = (float*)(ws + o);          o = up(o + (size_t)NN * FOP * 4);
  // total ~71 MB

  const int B = 256;
  auto cdiv = [](long long a, long long b) { return (int)((a + b - 1) / b); };

  // weights + norms
  gcn_prep_weights<<<cdiv(FH * KPAD1 + FOP * FH, B), B, 0, stream>>>(W1, W2, W1tH, W1tL, W2tH, W2tL);
  gcn_deg_init<<<cdiv(NN, B), B, 0, stream>>>(dinv);
  gcn_deg_accum<<<cdiv(NE, B), B, 0, stream>>>(ei, dinv);
  gcn_deg_finalize<<<cdiv(NN, B), B, 0, stream>>>(dinv);

  // layer 1
  gcn_gemm1_wmma<<<cdiv((long long)(NN / 16) * 32, B), B, 0, stream>>>(X, W1tH, W1tL, Hpre);
  gcn_zero4<<<cdiv((long long)NN * FH / 4, B), B, 0, stream>>>(Agg1, NN * FH / 4);
  gcn_scatter1<<<cdiv((long long)(NE + NN) * 32, B), B, 0, stream>>>(ei, dinv, Hpre, Agg1);
  gcn_bias_relu<<<cdiv((long long)NN * FH, B), B, 0, stream>>>(Agg1, b1);

  // layer 2
  gcn_gemm2_wmma<<<cdiv((long long)(NN / 16) * 32, B), B, 0, stream>>>(Agg1, W2tH, W2tL, H2);
  gcn_zero4<<<cdiv((long long)NN * FOP / 4, B), B, 0, stream>>>(Agg2, NN * FOP / 4);
  gcn_scatter2<<<cdiv((long long)(NE + NN) * 32, B), B, 0, stream>>>(ei, dinv, H2, Agg2);

  // output
  gcn_logsoftmax<<<cdiv((long long)NN * 32, B), B, 0, stream>>>(Agg2, b2, out);
}
